// AttentionPooler_82686710383188
// MI455X (gfx1250) — matrix operand order repaired and checked by reading, not compiled, not run on hardware
//
#include <hip/hip_runtime.h>
#include <hip/hip_bf16.h>
#include <stdint.h>

// ---------------- problem constants ----------------
#define B_    32
#define S_    4096
#define D_    768     // DIM
#define H_    12      // HEADS
#define DH_   64      // DIM_HEAD
#define NQ_   32      // NQUERY
#define NKV_  1536    // 2*INNER
#define SCH_  128     // s-chunk per workgroup
#define NCH_  (S_ / SCH_)          // 32 chunks
#define KSTEPS_ (D_ / 32)          // 24 k-steps of 32 for the KV GEMM
#define NTILES_KV_ (NKV_ / 16)     // 96 column tiles of Wkv

typedef unsigned short u16;
typedef unsigned int   u32;
typedef unsigned long long u64;
typedef __attribute__((ext_vector_type(16))) __bf16 v16bf;
typedef __attribute__((ext_vector_type(8)))  float  v8f;
typedef __attribute__((ext_vector_type(2)))  float  v2f;
typedef __attribute__((ext_vector_type(4)))  u32    v4u;
typedef __attribute__((ext_vector_type(8)))  int    v8i;
typedef __attribute__((ext_vector_type(4)))  int    v4i;

union Frag16 { uint4 q[2]; v16bf v; };   // one WMMA bf16 A/B fragment = 8 VGPRs

__device__ __forceinline__ u16 f2bf(float f) {
  u32 u = __float_as_uint(f);
  u32 r = u + 0x7FFFu + ((u >> 16) & 1u);   // round-to-nearest-even
  return (u16)(r >> 16);
}
__device__ __forceinline__ v8f v8f_zero() {
  v8f z = {0.f,0.f,0.f,0.f,0.f,0.f,0.f,0.f};
  return z;
}

// ---------------------------------------------------------------------------
// TDM: DMA one 4-tile B-fragment block (4 x 1024B rows, row stride = 24*1024B)
// from the pre-swizzled Wkv into LDS.  2-D descriptor, data_size = 8 bytes.
// 6-arg toolchain form: (g0, g1, g2, g3, extra, cpol); extra = zeros (no
// gather indices, non-cluster, 2-D tile).
// ---------------------------------------------------------------------------
__device__ __forceinline__ void tdm_load_block(const u16* gsrc, u32 lds_off) {
  u64 ga = (u64)(uintptr_t)gsrc;
  v4u g0;
  g0.x = 1u;                                   // count = 1 valid descriptor
  g0.y = lds_off;                              // LDS byte address
  g0.z = (u32)(ga & 0xFFFFFFFFu);              // global addr [31:0]
  g0.w = (u32)((ga >> 32) & 0x01FFFFFFu) | (2u << 30);  // [56:32] | type=2
  v8i g1;
  g1[0] = 0x00030000;          // wg_mask=0, data_size=3 (8B units)
  g1[1] = 0;                   // atomic_barrier_addr=0, tensor_dim0 lo16 = 0
  g1[2] = 0x4000;              // tensor_dim0 = 1<<30 (hi16), tensor_dim1 lo16=0
  g1[3] = (128 << 16) | 0x4000;// tensor_dim1 = 1<<30 (hi16), tile_dim0 = 128 units
  g1[4] = 4;                   // tile_dim1 = 4, tile_dim2 = 0 (2-D tile)
  g1[5] = 3072;                // tensor_dim0_stride = 3072 units (= 24*1024B)
  g1[6] = 0;                   // stride hi, dim1_stride lo = 0
  g1[7] = 0;
  v4i g2 = {0, 0, 0, 0};
  v4i g3 = {0, 0, 0, 0};
  v8i gx = {0, 0, 0, 0, 0, 0, 0, 0};
  __builtin_amdgcn_tensor_load_to_lds(g0, g1, g2, g3, gx, 0);
}

// ---------------------------------------------------------------------------
// K1a: convert Wkv (768 x 1536 f32) to bf16, pre-swizzled into WMMA B-fragment
// order: [ntile(96)][kstep(24)][lane(32)][16 bf16]   (contiguous k = 16*(lane/16) + e)
// ---------------------------------------------------------------------------
__global__ void k_prep_wkv(const float* __restrict__ Wkv, u16* __restrict__ swz) {
  int idx = blockIdx.x * blockDim.x + threadIdx.x;
  if (idx >= NTILES_KV_ * KSTEPS_ * 32 * 16) return;
  int e    = idx & 15;
  int rest = idx >> 4;
  int lane = rest & 31;
  rest >>= 5;
  int ks = rest % KSTEPS_;
  int nt = rest / KSTEPS_;
  int k  = ks * 32 + (lane >> 4) * 16 + e;
  int n  = nt * 16 + (lane & 15);
  swz[idx] = f2bf(Wkv[(size_t)k * NKV_ + n]);
}

// ---------------------------------------------------------------------------
// K1b: q = (queries @ Wq) * 1/sqrt(64), bf16, laid out [h][n][d]
// ---------------------------------------------------------------------------
__global__ void k_prep_q(const float* __restrict__ queries,
                         const float* __restrict__ Wq,
                         u16* __restrict__ qh) {
  int idx = blockIdx.x * blockDim.x + threadIdx.x;
  if (idx >= H_ * NQ_ * DH_) return;
  int d = idx & (DH_ - 1);
  int n = (idx >> 6) & (NQ_ - 1);
  int h = idx >> 11;
  int col = h * DH_ + d;
  float acc = 0.f;
  for (int k = 0; k < D_; ++k)
    acc += queries[n * D_ + k] * Wq[(size_t)k * D_ + col];
  qh[idx] = f2bf(acc * 0.125f);
}

// ---------------------------------------------------------------------------
// K2: fused LayerNorm + KV projection (TDM-fed) + flash partial attention.
// grid (NCH_, B_), 256 threads = 8 wave32.
// ---------------------------------------------------------------------------
__global__ void __launch_bounds__(256, 1)
k_attn_partial(const float* __restrict__ enc,
               const float* __restrict__ gamma,
               const float* __restrict__ beta,
               const u16*  __restrict__ swz,
               const u16*  __restrict__ qh,
               float* __restrict__ accp,
               float* __restrict__ mp,
               float* __restrict__ lp) {
  __shared__ u16   xln[SCH_][D_ + 8];      // LN'd activations, bf16  (194 KB)
  __shared__ u16   ktile[SCH_][DH_ + 8];   // K tile  [s][d]          (18 KB)
  __shared__ u16   vtileT[DH_][SCH_ + 8];  // V tile transposed [d][s](17 KB)
  __shared__ u16   ptile[NQ_][SCH_ + 8];   // softmax probs bf16      (8.5 KB)
  __shared__ __align__(32) u16 bstage[2][4096]; // TDM double buffer   (16 KB)
  __shared__ float redm[NQ_][16];
  __shared__ float reds[NQ_][16];
  __shared__ float rowm[NQ_];

  const int c   = blockIdx.x;
  const int b   = blockIdx.y;
  const int tid = threadIdx.x;
  const int w   = tid >> 5;     // wave id 0..7
  const int l   = tid & 31;     // lane
  const int lh  = l >> 4;       // half-wave 0/1
  const int l15 = l & 15;
  const int s0  = c * SCH_;

  // ---- Phase 1: LayerNorm, one wave per row, 16 rows per wave ----
  float4 gbuf[6], bbuf[6];
#pragma unroll
  for (int j = 0; j < 6; ++j) {
    gbuf[j] = *(const float4*)(gamma + j * 128 + l * 4);
    bbuf[j] = *(const float4*)(beta  + j * 128 + l * 4);
  }
  for (int rr = 0; rr < 16; ++rr) {
    int r = w * 16 + rr;
    const float* rowp = enc + ((size_t)b * S_ + (size_t)(s0 + r)) * D_;
    __builtin_prefetch(rowp + D_ + l * 24, 0, 3);   // next row -> global_prefetch_b8
    float4 buf[6];
    float s = 0.f, ss = 0.f;
#pragma unroll
    for (int j = 0; j < 6; ++j) {
      buf[j] = *(const float4*)(rowp + j * 128 + l * 4);
      s  += buf[j].x + buf[j].y + buf[j].z + buf[j].w;
      ss += buf[j].x*buf[j].x + buf[j].y*buf[j].y + buf[j].z*buf[j].z + buf[j].w*buf[j].w;
    }
#pragma unroll
    for (int m = 1; m < 32; m <<= 1) {
      s  += __shfl_xor(s,  m, 32);
      ss += __shfl_xor(ss, m, 32);
    }
    float mu   = s * (1.f / 768.f);
    float var  = ss * (1.f / 768.f) - mu * mu;
    float rstd = rsqrtf(var + 1e-5f);
#pragma unroll
    for (int j = 0; j < 6; ++j) {
      float x0 = (buf[j].x - mu) * rstd * gbuf[j].x + bbuf[j].x;
      float x1 = (buf[j].y - mu) * rstd * gbuf[j].y + bbuf[j].y;
      float x2 = (buf[j].z - mu) * rstd * gbuf[j].z + bbuf[j].z;
      float x3 = (buf[j].w - mu) * rstd * gbuf[j].w + bbuf[j].w;
      uint2 pk;
      pk.x = (u32)f2bf(x0) | ((u32)f2bf(x1) << 16);
      pk.y = (u32)f2bf(x2) | ((u32)f2bf(x3) << 16);
      *(uint2*)&xln[r][j * 128 + l * 4] = pk;
    }
  }

  // TDM prologue: stage (h=0, ks=0) K and V fragment blocks into slot 0
  const u32 bstage_lds = (u32)(uintptr_t)(void*)&bstage[0][0];
  if (w == 0) {
    tdm_load_block(swz + (size_t)(0 * KSTEPS_ + 0) * 512,        bstage_lds);
    tdm_load_block(swz + (size_t)(48 * KSTEPS_ + 0) * 512,       bstage_lds + 4096);
  }
  __syncthreads();   // xln visible to all waves

  for (int h = 0; h < H_; ++h) {
    // ---- 2a: K/V tile = xln(128x768) @ Wkv[:, head cols], TDM-fed WMMA ----
    v8f kacc[4], vacc[4];
#pragma unroll
    for (int t = 0; t < 4; ++t) { kacc[t] = v8f_zero(); vacc[t] = v8f_zero(); }
    const int am = w * 16 + l15;    // wave's A-matrix row
    for (int ks = 0; ks < KSTEPS_; ++ks) {
      const int cur = ks & 1;
      if (w == 0) __builtin_amdgcn_s_wait_tensorcnt(0);  // current slot ready
      __syncthreads();
      if (w == 0) {                  // kick DMA for the next stage
        const u32 nslot = bstage_lds + (u32)(1 - cur) * 8192;
        if (ks + 1 < KSTEPS_) {
          tdm_load_block(swz + ((size_t)(h * 4) * KSTEPS_ + (ks + 1)) * 512,        nslot);
          tdm_load_block(swz + ((size_t)(48 + h * 4) * KSTEPS_ + (ks + 1)) * 512,   nslot + 4096);
        } else if (h + 1 < H_) {
          tdm_load_block(swz + ((size_t)((h + 1) * 4) * KSTEPS_) * 512,             nslot);
          tdm_load_block(swz + ((size_t)(48 + (h + 1) * 4) * KSTEPS_) * 512,        nslot + 4096);
        }
      }
      const int kk = ks * 32;
      Frag16 A;   // 16x32 bf16 A: v0..3 from kk+16*lh, v4..7 from kk+16*lh+8
      A.q[0] = *(const uint4*)&xln[am][kk + lh * 16];
      A.q[1] = *(const uint4*)&xln[am][kk + lh * 16 + 8];
      const u16* bs = &bstage[cur][0];
#pragma unroll
      for (int t = 0; t < 4; ++t) {
        Frag16 Bk, Bv;
        const u16* bk = bs + t * 512 + l * 16;
        const u16* bv = bs + 2048 + t * 512 + l * 16;
        Bk.q[0] = *(const uint4*)(bk);
        Bk.q[1] = *(const uint4*)(bk + 8);
        Bv.q[0] = *(const uint4*)(bv);
        Bv.q[1] = *(const uint4*)(bv + 8);
        kacc[t] = __builtin_amdgcn_wmma_f32_16x16x32_bf16(false, A.v, false, Bk.v,
                                                          (short)0, kacc[t], false, false);
        vacc[t] = __builtin_amdgcn_wmma_f32_16x16x32_bf16(false, A.v, false, Bv.v,
                                                          (short)0, vacc[t], false, false);
      }
    }
#pragma unroll
    for (int t = 0; t < 4; ++t)
#pragma unroll
      for (int vi = 0; vi < 8; ++vi) {
        ktile [w * 16 + vi + 8 * lh][t * 16 + l15]  = f2bf(kacc[t][vi]);
        vtileT[t * 16 + l15][w * 16 + vi + 8 * lh]  = f2bf(vacc[t][vi]);
      }
    __syncthreads();

    // ---- 2b: scores = q(32x64) @ k^T, chunk-local softmax ----
    v8f sc0 = v8f_zero(), sc1 = v8f_zero();   // n=0..15 / n=16..31, s = w*16+l15
#pragma unroll
    for (int ks = 0; ks < 2; ++ks) {
      Frag16 Bf;   // B[k=d][n=s] = ktile[s][d]
      Bf.q[0] = *(const uint4*)&ktile[w * 16 + l15][ks * 32 + lh * 16];
      Bf.q[1] = *(const uint4*)&ktile[w * 16 + l15][ks * 32 + lh * 16 + 8];
      Frag16 A0, A1;
      const u16* q0 = qh + (size_t)(h * NQ_ + l15)      * DH_;
      const u16* q1 = qh + (size_t)(h * NQ_ + 16 + l15) * DH_;
      A0.q[0] = *(const uint4*)(q0 + ks * 32 + lh * 16);
      A0.q[1] = *(const uint4*)(q0 + ks * 32 + lh * 16 + 8);
      A1.q[0] = *(const uint4*)(q1 + ks * 32 + lh * 16);
      A1.q[1] = *(const uint4*)(q1 + ks * 32 + lh * 16 + 8);
      sc0 = __builtin_amdgcn_wmma_f32_16x16x32_bf16(false, A0.v, false, Bf.v,
                                                    (short)0, sc0, false, false);
      sc1 = __builtin_amdgcn_wmma_f32_16x16x32_bf16(false, A1.v, false, Bf.v,
                                                    (short)0, sc1, false, false);
    }
    // chunk max per query row n (s lives in lanes + waves)
    const int slot = w * 2 + lh;
#pragma unroll
    for (int vi = 0; vi < 8; ++vi) {
      float x0 = sc0[vi], x1 = sc1[vi];
#pragma unroll
      for (int m = 1; m < 16; m <<= 1) {
        x0 = fmaxf(x0, __shfl_xor(x0, m, 32));
        x1 = fmaxf(x1, __shfl_xor(x1, m, 32));
      }
      if (l15 == 0) {
        redm[vi + 8 * lh][slot]      = x0;
        redm[16 + vi + 8 * lh][slot] = x1;
      }
    }
    __syncthreads();
    if (tid < NQ_) {
      float mm = redm[tid][0];
#pragma unroll
      for (int j = 1; j < 16; ++j) mm = fmaxf(mm, redm[tid][j]);
      rowm[tid] = mm;
    }
    __syncthreads();
    // p = exp(score - m); write bf16 P tile; reduce row sums
#pragma unroll
    for (int vi = 0; vi < 8; ++vi) {
      float p0 = __expf(sc0[vi] - rowm[vi + 8 * lh]);
      float p1 = __expf(sc1[vi] - rowm[16 + vi + 8 * lh]);
      ptile[vi + 8 * lh][w * 16 + l15]      = f2bf(p0);
      ptile[16 + vi + 8 * lh][w * 16 + l15] = f2bf(p1);
#pragma unroll
      for (int m = 1; m < 16; m <<= 1) {
        p0 += __shfl_xor(p0, m, 32);
        p1 += __shfl_xor(p1, m, 32);
      }
      if (l15 == 0) {
        reds[vi + 8 * lh][slot]      = p0;
        reds[16 + vi + 8 * lh][slot] = p1;
      }
    }
    __syncthreads();
    if (tid < NQ_) {
      float sum = 0.f;
#pragma unroll
      for (int j = 0; j < 16; ++j) sum += reds[tid][j];
      size_t mi = (((size_t)b * NCH_ + c) * H_ + h) * NQ_ + tid;
      mp[mi] = rowm[tid];
      lp[mi] = sum;
    }
    __syncthreads();

    // ---- 2c: partial out = P(32x128) @ V(128x64), WMMA ----
    const int mt = w & 1, nt = w >> 1;    // 2 M-tiles x 4 N-tiles over 8 waves
    v8f oacc = v8f_zero();
#pragma unroll
    for (int ks = 0; ks < 4; ++ks) {
      Frag16 A, Bv2;
      A.q[0]   = *(const uint4*)&ptile[mt * 16 + l15][ks * 32 + lh * 16];
      A.q[1]   = *(const uint4*)&ptile[mt * 16 + l15][ks * 32 + lh * 16 + 8];
      Bv2.q[0] = *(const uint4*)&vtileT[nt * 16 + l15][ks * 32 + lh * 16];
      Bv2.q[1] = *(const uint4*)&vtileT[nt * 16 + l15][ks * 32 + lh * 16 + 8];
      oacc = __builtin_amdgcn_wmma_f32_16x16x32_bf16(false, A.v, false, Bv2.v,
                                                     (short)0, oacc, false, false);
    }
    float* ap = accp + ((((size_t)b * NCH_ + c) * H_ + h) * NQ_) * DH_;
#pragma unroll
    for (int vi = 0; vi < 8; ++vi)
      ap[(mt * 16 + vi + 8 * lh) * DH_ + nt * 16 + l15] = oacc[vi];
    __syncthreads();   // protect LDS tiles before next head
  }
}

// ---------------------------------------------------------------------------
// K3: flash-decoding combine over the 32 s-chunks; writes attn_out f32.
// grid (H_, B_), 256 threads.
// ---------------------------------------------------------------------------
__global__ void k_combine(const float* __restrict__ accp,
                          const float* __restrict__ mp,
                          const float* __restrict__ lp,
                          float* __restrict__ attn) {
  const int h = blockIdx.x, b = blockIdx.y, tid = threadIdx.x;
  __shared__ float coef[NCH_][NQ_];
  __shared__ float rcl[NQ_];
  if (tid < NQ_) {
    size_t base = (size_t)b * NCH_ * H_ * NQ_ + (size_t)h * NQ_ + tid;
    float M = -3.4e38f;
    for (int ci = 0; ci < NCH_; ++ci)
      M = fmaxf(M, mp[base + (size_t)ci * H_ * NQ_]);
    float L = 0.f;
    for (int ci = 0; ci < NCH_; ++ci) {
      float wc = __expf(mp[base + (size_t)ci * H_ * NQ_] - M);
      coef[ci][tid] = wc;
      L += lp[base + (size_t)ci * H_ * NQ_] * wc;
    }
    rcl[tid] = 1.f / L;
  }
  __syncthreads();
  for (int i = tid; i < NQ_ * DH_; i += 256) {
    int n = i >> 6, d = i & 63;
    float acc = 0.f;
    size_t abase = (((size_t)b * NCH_) * H_ + h) * (size_t)(NQ_ * DH_)
                 + (size_t)n * DH_ + d;
    for (int ci = 0; ci < NCH_; ++ci)
      acc += accp[abase + (size_t)ci * H_ * NQ_ * DH_] * coef[ci][n];
    attn[((size_t)b * NQ_ + n) * D_ + h * DH_ + d] = acc * rcl[n];
  }
}

// ---------------------------------------------------------------------------
// K4: out = attn_out (1024x768) @ Wo (768x768), full-precision f32 WMMA.
// 3072 16x16 tiles, 8 waves/block -> 384 blocks; K stepped 4 at a time.
// ---------------------------------------------------------------------------
__global__ void __launch_bounds__(256, 2)
k_outproj(const float* __restrict__ attn,
          const float* __restrict__ Wo,
          float* __restrict__ out) {
  const int tid = threadIdx.x;
  const int w = tid >> 5, l = tid & 31, lh = l >> 4, l15 = l & 15;
  const int tile = blockIdx.x * 8 + w;      // 0..3071
  const int mt = tile / 48, nt = tile % 48; // 64 M-tiles x 48 N-tiles
  v8f acc = v8f_zero();
  const float* arow = attn + (size_t)(mt * 16 + l15) * D_;
  const float* wcol = Wo + nt * 16 + l15;
  for (int kk = 0; kk < D_; kk += 4) {
    // A 16x4 f32: lane m=l15, k = 2*lh + v  -> contiguous float2
    const float2 av = *(const float2*)(arow + kk + lh * 2);
    v2f a; a.x = av.x; a.y = av.y;
    // B 4x16 f32: lane n=l15, k = 2*lh + v  -> two strided b32 loads
    v2f bb;
    bb.x = wcol[(size_t)(kk + lh * 2) * D_];
    bb.y = wcol[(size_t)(kk + lh * 2 + 1) * D_];
    acc = __builtin_amdgcn_wmma_f32_16x16x4_f32(false, a, false, bb,
                                                (short)0, acc, false, false);
  }
#pragma unroll
  for (int vi = 0; vi < 8; ++vi)
    out[(size_t)(mt * 16 + vi + 8 * lh) * D_ + nt * 16 + l15] = acc[vi];
}

// ---------------------------------------------------------------------------
extern "C" void kernel_launch(void* const* d_in, const int* in_sizes, int n_in,
                              void* d_out, int out_size, void* d_ws, size_t ws_size,
                              hipStream_t stream) {
  (void)in_sizes; (void)n_in; (void)out_size; (void)ws_size;
  const float* enc     = (const float*)d_in[0];
  const float* queries = (const float*)d_in[1];
  const float* Wq      = (const float*)d_in[2];
  const float* Wkv     = (const float*)d_in[3];
  const float* Wo      = (const float*)d_in[4];
  const float* gamma   = (const float*)d_in[5];
  const float* beta    = (const float*)d_in[6];
  float* out = (float*)d_out;

  char* ws = (char*)d_ws;
  const size_t SZ_SWZ = (size_t)NTILES_KV_ * KSTEPS_ * 32 * 16 * sizeof(u16); // 2.36 MB
  const size_t SZ_QH  = (size_t)H_ * NQ_ * DH_ * sizeof(u16);                 // 48 KB
  const size_t SZ_ACC = (size_t)B_ * NCH_ * H_ * NQ_ * DH_ * sizeof(float);   // 100.7 MB
  const size_t SZ_ML  = (size_t)B_ * NCH_ * H_ * NQ_ * sizeof(float);         // 1.57 MB
  const size_t OFF_SWZ = 0;
  const size_t OFF_QH  = OFF_SWZ + SZ_SWZ;
  const size_t OFF_ACC = OFF_QH + SZ_QH;
  const size_t OFF_M   = OFF_ACC + SZ_ACC;
  const size_t OFF_L   = OFF_M + SZ_ML;
  const size_t OFF_ATT = OFF_L + SZ_ML;

  u16*  swz  = (u16*) (ws + OFF_SWZ);
  u16*  qh   = (u16*) (ws + OFF_QH);
  float* accp = (float*)(ws + OFF_ACC);
  float* mpv  = (float*)(ws + OFF_M);
  float* lpv  = (float*)(ws + OFF_L);
  float* attn = (float*)(ws + OFF_ATT);

  k_prep_wkv<<<(NTILES_KV_*KSTEPS_*32*16 + 255)/256, 256, 0, stream>>>(Wkv, swz);
  k_prep_q  <<<(H_*NQ_*DH_ + 255)/256,               256, 0, stream>>>(queries, Wq, qh);
  k_attn_partial<<<dim3(NCH_, B_), 256, 0, stream>>>(enc, gamma, beta, swz, qh,
                                                     accp, mpv, lpv);
  k_combine <<<dim3(H_, B_), 256, 0, stream>>>(accp, mpv, lpv, attn);
  k_outproj <<<B_ * NQ_, 256, 0, stream>>>(attn, Wo, out);
}
